// MultiModalGAT_29583734734923
// MI455X (gfx1250) — matrix operand order repaired and checked by reading, not compiled, not run on hardware
//
#include <hip/hip_runtime.h>
#include <hip/hip_bf16.h>

typedef __attribute__((ext_vector_type(16))) _Float16 v16h;
typedef __attribute__((ext_vector_type(8)))  _Float16 v8h;
typedef __attribute__((ext_vector_type(8)))  float    v8f;
typedef __attribute__((ext_vector_type(4)))  float    v4f;
typedef __attribute__((ext_vector_type(4)))  int      v4i;

enum { ACT_NONE = 0, ACT_RELU = 1, ACT_ELU = 2 };

#if defined(__has_builtin)
#if __has_builtin(__builtin_amdgcn_global_load_async_to_lds_b128)
#define ASYNC_LDS 1
#endif
#endif

__device__ __forceinline__ void wait_asynccnt0()
{
#if defined(__has_builtin)
#if __has_builtin(__builtin_amdgcn_s_wait_asynccnt)
    __builtin_amdgcn_s_wait_asynccnt(0);
    return;
#else
    asm volatile("s_wait_asynccnt 0" ::: "memory");
#endif
#else
    asm volatile("s_wait_asynccnt 0" ::: "memory");
#endif
}

// ---------------------------------------------------------------------------
// WMMA GEMM:  C[M,Nc] = act(A[M,K] @ W[Nc,K]^T + bias)
//
// Workgroup = 8 waves = 8 M-tiles sharing one 128-wide N-block. The weight
// panel (128 rows x 64 K) is double-buffered in LDS (padded stride vs. bank
// conflicts) and software-pipelined against 16 v_wmma_f32_16x16x32_f16 per
// panel. Panel fill uses gfx1250 async global->LDS DMA when available.
// ---------------------------------------------------------------------------
constexpr int KP   = 64;   // K-panel size
constexpr int PSTR = 72;   // padded LDS row stride (halves): 144B rotates banks

template<int ACT, bool A_IS_F32, bool OUT_F16>
__global__ __launch_bounds__(256)
void wmma_gemm_kernel(const void* __restrict__ Ap,
                      const _Float16* __restrict__ W,
                      const float* __restrict__ bias,
                      void* __restrict__ Cp,
                      int M, int Nc, int K, int lda, int ldc, int coloff)
{
    __shared__ _Float16 sB[2][128 * PSTR];          // 2 x 18 KB

    const int nblocks = Nc >> 7;                     // Nc / 128
    const int mg  = blockIdx.x / nblocks;            // group of 8 M-tiles
    const int nb  = blockIdx.x - mg * nblocks;       // N-block
    const int wv  = threadIdx.x >> 5;                // wave -> M-tile in group
    const int Mtiles = M >> 4;
    int tm = mg * 8 + wv;
    const bool valid = (tm < Mtiles);
    if (!valid) tm = Mtiles - 1;                     // clamp: keep barrier-uniform

    const int tid  = threadIdx.x;
    const int lane = threadIdx.x & 31;
    const int r    = lane & 15;                      // A-row / B-row / C-col in tile
    const int hi   = lane >> 4;                      // K-half selector

    const size_t arow = (size_t)(tm * 16 + r) * (size_t)lda;
    const _Float16* wpanel = W + (size_t)(nb * 128) * (size_t)K;

    v8f acc[8];
#pragma unroll
    for (int t = 0; t < 8; ++t) acc[t] = (v8f){};

    // ---- panel fill: 128 rows x 64 halves = 1024 x 16B chunks, 4 per thread ----
#ifndef ASYNC_LDS
    v8h fr[4];
#endif
    auto issue_fill = [&](int buf, int k0) {
#pragma unroll
        for (int q = 0; q < 4; ++q) {
            const int idx = q * 256 + tid;
            const int row = idx >> 3;
            const int ch  = idx & 7;
            const _Float16* src = wpanel + (size_t)row * K + k0 + ch * 8;
#ifdef ASYNC_LDS
            _Float16* dst = &sB[buf][row * PSTR + ch * 8];
            __builtin_amdgcn_global_load_async_to_lds_b128(
                (__attribute__((address_space(1))) v4i*)src,
                (__attribute__((address_space(3))) v4i*)dst,
                0, 0);
#else
            fr[q] = *(const v8h*)src;
#endif
        }
    };
    auto commit_fill = [&](int buf) {
#ifdef ASYNC_LDS
        (void)buf;
        wait_asynccnt0();
#else
#pragma unroll
        for (int q = 0; q < 4; ++q) {
            const int idx = q * 256 + tid;
            const int row = idx >> 3;
            const int ch  = idx & 7;
            *(v8h*)&sB[buf][row * PSTR + ch * 8] = fr[q];
        }
#endif
    };

    // prologue: panel 0
    issue_fill(0, 0);
    commit_fill(0);
    __syncthreads();

    int buf = 0;
    for (int kp = 0; kp < K; kp += KP, buf ^= 1) {
        const bool more = (kp + KP) < K;
        if (more) issue_fill(buf ^ 1, kp + KP);      // overlaps compute below

#pragma unroll
        for (int kc = 0; kc < KP; kc += 32) {
            const int kA = kp + kc;
            v16h a;
            if (A_IS_F32) {
                const float* p0 = (const float*)Ap + arow + kA + hi * 8;
                v4f f0 = *(const v4f*)(p0);
                v4f f1 = *(const v4f*)(p0 + 4);
                v4f f2 = *(const v4f*)(p0 + 16);
                v4f f3 = *(const v4f*)(p0 + 20);
#pragma unroll
                for (int i = 0; i < 4; ++i) {
                    a[i]      = (_Float16)f0[i];
                    a[4 + i]  = (_Float16)f1[i];
                    a[8 + i]  = (_Float16)f2[i];
                    a[12 + i] = (_Float16)f3[i];
                }
            } else {
                const _Float16* p0 = (const _Float16*)Ap + arow + kA + hi * 8;
                v8h h0 = *(const v8h*)(p0);
                v8h h1 = *(const v8h*)(p0 + 16);
#pragma unroll
                for (int i = 0; i < 8; ++i) { a[i] = h0[i]; a[8 + i] = h1[i]; }
            }
#pragma unroll
            for (int t = 0; t < 8; ++t) {
                const _Float16* bp = &sB[buf][(t * 16 + r) * PSTR + kc + hi * 8];
                v8h b0 = *(const v8h*)(bp);
                v8h b1 = *(const v8h*)(bp + 16);
                v16h b;
#pragma unroll
                for (int i = 0; i < 8; ++i) { b[i] = b0[i]; b[8 + i] = b1[i]; }
                acc[t] = __builtin_amdgcn_wmma_f32_16x16x32_f16(
                    /*neg_a=*/false, a, /*neg_b=*/false, b,
                    /*c_mod=*/(short)0, acc[t], /*reuse_a=*/false, /*reuse_b=*/false);
            }
        }

        if (more) commit_fill(buf ^ 1);
        __syncthreads();
    }

    if (!valid) return;                              // tail waves skip stores only
#pragma unroll
    for (int t = 0; t < 8; ++t) {
        const int col  = coloff + (nb * 8 + t) * 16 + r;
        const float bv = bias ? bias[(nb * 8 + t) * 16 + r] : 0.0f;
#pragma unroll
        for (int i = 0; i < 8; ++i) {
            float v = acc[t][i] + bv;
            if (ACT == ACT_RELU) v = fmaxf(v, 0.0f);
            if (ACT == ACT_ELU)  v = (v > 0.0f) ? v : (expf(v) - 1.0f);
            const size_t row = (size_t)(tm * 16 + hi * 8 + i);
            if (OUT_F16) ((_Float16*)Cp)[row * (size_t)ldc + col] = (_Float16)v;
            else         ((float*)Cp)[row * (size_t)ldc + col]    = v;
        }
    }
}

// ---------------------------------------------------------------------------
// Small helper kernels
// ---------------------------------------------------------------------------
__global__ void cvt_f32_f16(const float* __restrict__ in, _Float16* __restrict__ out, int n)
{
    int i = blockIdx.x * blockDim.x + threadIdx.x;
    if (i < n) out[i] = (_Float16)in[i];
}

__global__ void cvt_text_to_xcat(const float* __restrict__ text, _Float16* __restrict__ xcat, int total)
{
    int i = blockIdx.x * blockDim.x + threadIdx.x;
    if (i >= total) return;
    int row = i / 768;
    int c   = i - row * 768;
    xcat[(size_t)row * 896 + c] = (_Float16)text[i];
}

__global__ void fill_bias(float* __restrict__ dst, const float* __restrict__ b, int n, int mask)
{
    int i = blockIdx.x * blockDim.x + threadIdx.x;
    if (i < n) dst[i] = b[i & mask];
}

__global__ void elu_inplace(float* __restrict__ x, int n)
{
    int i = blockIdx.x * blockDim.x + threadIdx.x;
    if (i < n) { float v = x[i]; x[i] = (v > 0.0f) ? v : (expf(v) - 1.0f); }
}

__global__ void attn1_kernel(const float* __restrict__ xp,
                             const float* __restrict__ a_s, const float* __restrict__ a_d,
                             float* __restrict__ as1, float* __restrict__ ad1, int N)
{
    int i = blockIdx.x * blockDim.x + threadIdx.x;
    if (i >= N) return;
    const float* row = xp + (size_t)i * 256;
    float s0 = 0, d0 = 0, s1 = 0, d1 = 0;
    for (int c = 0; c < 128; ++c) {
        float v = row[c];       s0 += v * a_s[c];       d0 += v * a_d[c];
        float w = row[128 + c]; s1 += w * a_s[128 + c]; d1 += w * a_d[128 + c];
    }
    as1[i * 2] = s0; as1[i * 2 + 1] = s1;
    ad1[i * 2] = d0; ad1[i * 2 + 1] = d1;
}

__device__ __forceinline__ unsigned enc_f(float f)
{
    unsigned u = __float_as_uint(f);
    return (u & 0x80000000u) ? ~u : (u | 0x80000000u);
}
__device__ __forceinline__ float dec_f(unsigned u)
{
    return (u & 0x80000000u) ? __uint_as_float(u & 0x7FFFFFFFu) : __uint_as_float(~u);
}

__device__ __forceinline__ void edge_sd(const long long* ei, int E, int e, int& s, int& d)
{
    if (e < E) { s = (int)ei[e]; d = (int)ei[E + e]; }
    else       { s = e - E; d = e - E; }            // self-loops appended
}

__global__ void edge_max(const long long* __restrict__ ei, int E, int ET,
                         const float* __restrict__ as, const float* __restrict__ ad,
                         unsigned* __restrict__ mbuf, int H)
{
    int e = blockIdx.x * blockDim.x + threadIdx.x;
    if (e >= ET) return;
    int s, d; edge_sd(ei, E, e, s, d);
    for (int h = 0; h < H; ++h) {
        float v = as[s * H + h] + ad[d * H + h];
        v = (v >= 0.0f) ? v : 0.2f * v;              // leaky_relu, slope 0.2
        atomicMax(&mbuf[d * H + h], enc_f(v));
    }
}

__global__ void edge_expsum(const long long* __restrict__ ei, int E, int ET,
                            const float* __restrict__ as, const float* __restrict__ ad,
                            const unsigned* __restrict__ mbuf,
                            float* __restrict__ ex, float* __restrict__ den, int H)
{
    int e = blockIdx.x * blockDim.x + threadIdx.x;
    if (e >= ET) return;
    int s, d; edge_sd(ei, E, e, s, d);
    for (int h = 0; h < H; ++h) {
        float v = as[s * H + h] + ad[d * H + h];
        v = (v >= 0.0f) ? v : 0.2f * v;
        float w = expf(v - dec_f(mbuf[d * H + h]));
        ex[(size_t)e * H + h] = w;
        atomicAdd(&den[d * H + h], w);
    }
}

// GAT1 aggregate: one wave per edge, 256 channels (2 heads x 128)
__global__ void edge_agg1(const long long* __restrict__ ei, int E, int ET,
                          const float* __restrict__ xp,
                          const float* __restrict__ ex, const float* __restrict__ den,
                          float* __restrict__ out)
{
    int e = blockIdx.x * (blockDim.x >> 5) + (threadIdx.x >> 5);
    if (e >= ET) return;
    int lane = threadIdx.x & 31;
    int s, d; edge_sd(ei, E, e, s, d);
    float a0 = ex[(size_t)e * 2]     / den[d * 2];
    float a1 = ex[(size_t)e * 2 + 1] / den[d * 2 + 1];
    const float* xr = xp + (size_t)s * 256;
    float* orow = out + (size_t)d * 256;
#pragma unroll
    for (int j0 = 0; j0 < 256; j0 += 32) {
        int j = j0 + lane;
        float al = (j < 128) ? a0 : a1;
        atomicAdd(&orow[j], xr[j] * al);
    }
}

// tiny GEMM4 (256 -> 4) fused with second-layer attention dots
__global__ void gemm4_attn(const float* __restrict__ x1, const float* __restrict__ W2,
                           const float* __restrict__ a_s, const float* __restrict__ a_d,
                           float* __restrict__ xp2, float* __restrict__ as2,
                           float* __restrict__ ad2, int N)
{
    int i = blockIdx.x * blockDim.x + threadIdx.x;
    if (i >= N) return;
    const float* row = x1 + (size_t)i * 256;
    float a0 = 0, a1 = 0, a2 = 0, a3 = 0;
    for (int k = 0; k < 256; ++k) {
        float v = row[k];
        a0 += v * W2[k];
        a1 += v * W2[256 + k];
        a2 += v * W2[512 + k];
        a3 += v * W2[768 + k];
    }
    xp2[i * 4]     = a0; xp2[i * 4 + 1] = a1;
    xp2[i * 4 + 2] = a2; xp2[i * 4 + 3] = a3;
    as2[i] = a0 * a_s[0] + a1 * a_s[1] + a2 * a_s[2] + a3 * a_s[3];
    ad2[i] = a0 * a_d[0] + a1 * a_d[1] + a2 * a_d[2] + a3 * a_d[3];
}

__global__ void edge_agg2(const long long* __restrict__ ei, int E, int ET,
                          const float* __restrict__ xp2,
                          const float* __restrict__ ex, const float* __restrict__ den,
                          float* __restrict__ out)
{
    int e = blockIdx.x * blockDim.x + threadIdx.x;
    if (e >= ET) return;
    int s, d; edge_sd(ei, E, e, s, d);
    float al = ex[e] / den[d];
#pragma unroll
    for (int c = 0; c < 4; ++c)
        atomicAdd(&out[(size_t)d * 4 + c], xp2[(size_t)s * 4 + c] * al);
}

// ---------------------------------------------------------------------------
extern "C" void kernel_launch(void* const* d_in, const int* in_sizes, int n_in,
                              void* d_out, int out_size, void* d_ws, size_t ws_size,
                              hipStream_t stream)
{
    const float*     text = (const float*)d_in[0];
    const float*     mel  = (const float*)d_in[1];
    const long long* ei   = (const long long*)d_in[2];
    const float* Wm = (const float*)d_in[3];  const float* bm = (const float*)d_in[4];
    const float* Wc = (const float*)d_in[5];  const float* bc = (const float*)d_in[6];
    const float* W1 = (const float*)d_in[7];
    const float* as1w = (const float*)d_in[8];  const float* ad1w = (const float*)d_in[9];
    const float* b1 = (const float*)d_in[10];
    const float* W2 = (const float*)d_in[11];
    const float* as2w = (const float*)d_in[12]; const float* ad2w = (const float*)d_in[13];
    const float* b2 = (const float*)d_in[14];

    const int N  = in_sizes[0] / 768;
    const int E  = in_sizes[2] / 2;
    const int ET = E + N;

    // ---- workspace layout (256B aligned) ----
    char* w = (char*)d_ws;
    size_t o = 0;
    auto take = [&](size_t bytes) { size_t r = o; o += (bytes + 255) & ~(size_t)255; return r; };
    size_t oXP   = take((size_t)N * 256 * 4);   // xp  [N,256] f32
    size_t oXCAT = take((size_t)N * 896 * 2);   // xcat[N,896] f16 ; reused as out1/x1 [N,256] f32
    size_t oX16  = take((size_t)N * 128 * 2);   // x   [N,128] f16
    size_t oAS1  = take((size_t)N * 2 * 4);
    size_t oAD1  = take((size_t)N * 2 * 4);
    size_t oM1   = take((size_t)N * 2 * 4);
    size_t oD1   = take((size_t)N * 2 * 4);
    size_t oEX1  = take((size_t)ET * 2 * 4);
    size_t oXP2  = take((size_t)N * 4 * 4);
    size_t oAS2  = take((size_t)N * 4);
    size_t oAD2  = take((size_t)N * 4);
    size_t oM2   = take((size_t)N * 4);
    size_t oD2   = take((size_t)N * 4);
    size_t oEX2  = take((size_t)ET * 4);
    size_t oWM   = take((size_t)128 * 1024 * 2);
    size_t oWC   = take((size_t)128 * 896 * 2);
    size_t oW1   = take((size_t)256 * 128 * 2);

    float*     xp   = (float*)(w + oXP);
    _Float16*  xcat = (_Float16*)(w + oXCAT);
    float*     out1 = (float*)(w + oXCAT);     // alias: xcat dead after GEMM2
    _Float16*  x16  = (_Float16*)(w + oX16);
    float*     as1  = (float*)(w + oAS1);
    float*     ad1  = (float*)(w + oAD1);
    unsigned*  m1   = (unsigned*)(w + oM1);
    float*     den1 = (float*)(w + oD1);
    float*     ex1  = (float*)(w + oEX1);
    float*     xp2  = (float*)(w + oXP2);
    float*     as2  = (float*)(w + oAS2);
    float*     ad2  = (float*)(w + oAD2);
    unsigned*  m2   = (unsigned*)(w + oM2);
    float*     den2 = (float*)(w + oD2);
    float*     ex2  = (float*)(w + oEX2);
    _Float16*  Wm16 = (_Float16*)(w + oWM);
    _Float16*  Wc16 = (_Float16*)(w + oWC);
    _Float16*  W116 = (_Float16*)(w + oW1);

    const int T = 256;
    auto blks = [](int n, int t) { return (n + t - 1) / t; };

    // ---- weight / input conversion to f16 ----
    cvt_f32_f16<<<blks(128 * 1024, T), T, 0, stream>>>(Wm, Wm16, 128 * 1024);
    cvt_f32_f16<<<blks(128 * 896, T), T, 0, stream>>>(Wc, Wc16, 128 * 896);
    cvt_f32_f16<<<blks(256 * 128, T), T, 0, stream>>>(W1, W116, 256 * 128);
    cvt_text_to_xcat<<<blks(N * 768, T), T, 0, stream>>>(text, xcat, N * 768);

    const int Mtiles  = N / 16;
    const int mgroups = (Mtiles + 7) / 8;

    // ---- GEMM1: m = relu(mel @ Wm^T + bm) -> xcat cols [768,896) ----
    wmma_gemm_kernel<ACT_RELU, true, true><<<mgroups * 1, T, 0, stream>>>(
        mel, Wm16, bm, xcat, N, 128, 1024, 1024, 896, 768);
    // ---- GEMM2: x = elu(xcat @ Wc^T + bc) -> x16 [N,128] ----
    wmma_gemm_kernel<ACT_ELU, false, true><<<mgroups * 1, T, 0, stream>>>(
        xcat, Wc16, bc, x16, N, 128, 896, 896, 128, 0);
    // ---- GEMM3: xp = x @ W1^T -> fp32 [N,256] (bias added post-aggregation) ----
    wmma_gemm_kernel<ACT_NONE, false, false><<<mgroups * 2, T, 0, stream>>>(
        x16, W116, nullptr, xp, N, 256, 128, 128, 256, 0);

    // ---- GAT layer 1 ----
    attn1_kernel<<<blks(N, T), T, 0, stream>>>(xp, as1w, ad1w, as1, ad1, N);
    (void)hipMemsetAsync(m1, 0, (size_t)N * 2 * 4, stream);   // enc(-max)
    (void)hipMemsetAsync(den1, 0, (size_t)N * 2 * 4, stream);
    edge_max<<<blks(ET, T), T, 0, stream>>>(ei, E, ET, as1, ad1, m1, 2);
    edge_expsum<<<blks(ET, T), T, 0, stream>>>(ei, E, ET, as1, ad1, m1, ex1, den1, 2);
    fill_bias<<<blks(N * 256, T), T, 0, stream>>>(out1, b1, N * 256, 255);  // out1 = b1 (xcat now dead)
    edge_agg1<<<blks(ET, 8), T, 0, stream>>>(ei, E, ET, xp, ex1, den1, out1);
    elu_inplace<<<blks(N * 256, T), T, 0, stream>>>(out1, N * 256);         // x1 = elu(out1)

    // ---- GAT layer 2 (1 head, 4 channels) ----
    gemm4_attn<<<blks(N, T), T, 0, stream>>>(out1, W2, as2w, ad2w, xp2, as2, ad2, N);
    (void)hipMemsetAsync(m2, 0, (size_t)N * 4, stream);
    (void)hipMemsetAsync(den2, 0, (size_t)N * 4, stream);
    edge_max<<<blks(ET, T), T, 0, stream>>>(ei, E, ET, as2, ad2, m2, 1);
    edge_expsum<<<blks(ET, T), T, 0, stream>>>(ei, E, ET, as2, ad2, m2, ex2, den2, 1);
    fill_bias<<<blks(N * 4, T), T, 0, stream>>>((float*)d_out, b2, N * 4, 3);
    edge_agg2<<<blks(ET, T), T, 0, stream>>>(ei, E, ET, xp2, ex2, den2, (float*)d_out);

    (void)n_in; (void)out_size; (void)ws_size;
}
